// ProposalToDetectBox_26800595927042
// MI455X (gfx1250) — compile-verified, matrix operand order checked by reading, not run on hardware
//
#include <hip/hip_runtime.h>
#include <hip/hip_bf16.h>

typedef __attribute__((ext_vector_type(16))) _Float16 v16h;
typedef __attribute__((ext_vector_type(8)))  float    v8f;

#define BB 32
#define NN 8000
#define CC 81
#define KK 100
#define SCORE_THR 0.7f
#define IOU_THR   0.3f

// ---------------------------------------------------------------------------
// Kernel 0: zero the per-image candidate counters (ws is poisoned, not zeroed)
// ---------------------------------------------------------------------------
__global__ void init_counts_kernel(int* __restrict__ ccount) {
  if (threadIdx.x < BB) ccount[threadIdx.x] = 0;
}

// ---------------------------------------------------------------------------
// Kernel 1: score + regress + compact.
// One wave handles 16 proposals. Softmax denominator via WMMA row-sum:
//   A[16x32] = exp(logit - max) chunks (f16), B = ones -> D rows are sums.
// A-matrix f16 16x32 layout (ISA 7.12.2): lane l holds row M=l%16;
//   lanes 0-15: elements 0..7 -> K=0..7,  8..15 -> K=16..23
//   lanes 16-31: elements 0..7 -> K=8..15, 8..15 -> K=24..31
// ---------------------------------------------------------------------------
__global__ __launch_bounds__(256) void score_compact_kernel(
    const float* __restrict__ deltas,     // [B,N,4]
    const float* __restrict__ logits,     // [B,N,81]
    const float* __restrict__ props,      // [B,N,5]
    float* __restrict__ cy1, float* __restrict__ cx1,
    float* __restrict__ cy2, float* __restrict__ cx2,
    float* __restrict__ cscore,
    int* __restrict__ ccls, int* __restrict__ cidx,
    int* __restrict__ ccount)
{
  const int lane = threadIdx.x & 31;
  const int q    = lane & 15;           // proposal slot within wave
  const int h    = lane >> 4;           // which half of the K interleave
  const long long wave = (long long)blockIdx.x * (blockDim.x >> 5) + (threadIdx.x >> 5);
  const long long g    = wave * 16 + q;            // global proposal id, exact cover
  const int b = (int)(g / NN);
  const int n = (int)(g % NN);
  const float* __restrict__ row = logits + (size_t)g * CC;

  // ---- load this lane's 48 class logits (3 chunks x 16), mask classes >= 81
  float lg[48];
#pragma unroll
  for (int c = 0; c < 3; ++c) {
#pragma unroll
    for (int e = 0; e < 16; ++e) {
      const int cls = 32 * c + ((e & 8) ? 16 : 0) + 8 * h + (e & 7);
      lg[c * 16 + e] = (cls < CC) ? row[cls] : -__builtin_inff();
    }
  }

  // ---- max over all classes; max+argmax over classes 1..80 (first-index tie)
  float mall = -__builtin_inff();
  float mc   = -__builtin_inff();
  int   ac   = 1 << 30;
#pragma unroll
  for (int c = 0; c < 3; ++c) {
#pragma unroll
    for (int e = 0; e < 16; ++e) {
      const int cls = 32 * c + ((e & 8) ? 16 : 0) + 8 * h + (e & 7);
      const float v = lg[c * 16 + e];
      mall = fmaxf(mall, v);
      if (cls >= 1 && cls < CC) {
        if (v > mc || (v == mc && cls < ac)) { mc = v; ac = cls; }
      }
    }
  }
  // combine the two lanes (l and l^16) that share one proposal
  {
    const float om = __shfl_xor(mall, 16);
    mall = fmaxf(mall, om);
    const float omc = __shfl_xor(mc, 16);
    const int   oac = __shfl_xor(ac, 16);
    if (omc > mc || (omc == mc && oac < ac)) { mc = omc; ac = oac; }
  }

  // ---- sum of exp via WMMA: acc = exp(A) x Ones  (3 K-chunks of 32)
  v8f acc = {};
  v16h ones;
#pragma unroll
  for (int e = 0; e < 16; ++e) ones[e] = (_Float16)1.0f;
#pragma unroll
  for (int c = 0; c < 3; ++c) {
    v16h a;
#pragma unroll
    for (int e = 0; e < 16; ++e)
      a[e] = (_Float16)__expf(lg[c * 16 + e] - mall);   // <=1.0, f16-safe; -inf -> 0
    acc = __builtin_amdgcn_wmma_f32_16x16x32_f16(false, a, false, ones,
                                                 (short)0, acc, false, false);
  }
  // D layout: lanes 0-15 VGPR r = row M=r; lanes 16-31 VGPR r = row M=r+8.
  // Proposal q's sum lives in acc[q&7] of lane q (q<8) or lane q+16 (q>=8).
  float eacc = acc[0];
#pragma unroll
  for (int r = 1; r < 8; ++r)
    if ((q & 7) == r) eacc = acc[r];
  const float sum = __shfl(eacc, (q < 8) ? q : (q + 16));

  // ---- threshold, regress, compact (one writer lane per proposal)
  if (h == 0) {
    const float score = __expf(mc - mall) / sum;
    if (score >= SCORE_THR) {
      const float* dp = deltas + (size_t)g * 4;
      const float* pp = props  + (size_t)g * 5;
      const float y1 = pp[0], x1 = pp[1], y2 = pp[2], x2 = pp[3];
      float bh = y2 - y1, bw = x2 - x1;
      const float cy = (y2 + y1) * 0.5f + dp[0] * bh;
      const float cx = (x2 + x1) * 0.5f + dp[1] * bw;
      bh *= __expf(dp[2]);
      bw *= __expf(dp[3]);
      const int pos = atomicAdd(&ccount[b], 1);
      const size_t o = (size_t)b * NN + (size_t)pos;
      cy1[o] = cy - 0.5f * bh;
      cx1[o] = cx - 0.5f * bw;
      cy2[o] = cy + 0.5f * bh;
      cx2[o] = cx + 0.5f * bw;
      cscore[o] = score;
      ccls[o]   = ac;
      cidx[o]   = n;
    }
  }
}

// ---------------------------------------------------------------------------
// Kernel 2: class-aware greedy NMS, one 256-thread workgroup per image.
// Scores live in LDS; candidate boxes stay L2-resident (192MB L2 >> 7MB ws).
// Same-class IoU check == reference's per-class coordinate-offset trick.
// ---------------------------------------------------------------------------
__global__ __launch_bounds__(256) void nms_kernel(
    const float* __restrict__ logits,     // [B,N,81] original logits
    const float* __restrict__ cy1, const float* __restrict__ cx1,
    const float* __restrict__ cy2, const float* __restrict__ cx2,
    const float* __restrict__ cscore,
    const int* __restrict__ ccls, const int* __restrict__ cidx,
    const int* __restrict__ ccount,
    float* __restrict__ out)
{
  __shared__ float         sc[NN];
  __shared__ unsigned char scls8[NN];
  __shared__ float redv[8];
  __shared__ int   redi[8];
  __shared__ float jinfo[6];   // y1,x1,y2,x2,area,score
  __shared__ int   jint[3];    // cls, cand index, original proposal index

  const int b = blockIdx.x;
  const int t = threadIdx.x;
  int M = ccount[b];
  if (M > NN) M = NN;
  const size_t base = (size_t)b * NN;

  for (int i = t; i < M; i += 256) {
    sc[i]    = cscore[base + i];
    scls8[i] = (unsigned char)ccls[base + i];
  }
  __syncthreads();

  float* __restrict__ oboxes  = out;
  float* __restrict__ oscores = out + (size_t)BB * KK * 5;
  float* __restrict__ oclsout = oscores + (size_t)BB * KK * 2;
  float* __restrict__ ologits = oclsout + (size_t)BB * KK * 2;

  for (int k = 0; k < KK; ++k) {
    // ---- block argmax (value desc, index asc on ties)
    float bv = -__builtin_inff();
    int   bi = 0x7fffffff;
    for (int i = t; i < M; i += 256) {
      const float v = sc[i];
      if (v > bv || (v == bv && i < bi)) { bv = v; bi = i; }
    }
#pragma unroll
    for (int off = 16; off; off >>= 1) {
      const float ov = __shfl_xor(bv, off);
      const int   oi = __shfl_xor(bi, off);
      if (ov > bv || (ov == bv && oi < bi)) { bv = ov; bi = oi; }
    }
    if ((t & 31) == 0) { redv[t >> 5] = bv; redi[t >> 5] = bi; }
    __syncthreads();
    if (t == 0) {
      float fv = redv[0];
      int   fi = redi[0];
#pragma unroll
      for (int w = 1; w < 8; ++w) {
        const float ov = redv[w];
        const int   oi = redi[w];
        if (ov > fv || (ov == fv && oi < fi)) { fv = ov; fi = oi; }
      }
      if (fv > -__builtin_inff()) {
        const size_t o = base + (size_t)fi;
        const float y1 = cy1[o], x1 = cx1[o], y2 = cy2[o], x2 = cx2[o];
        jinfo[0] = y1; jinfo[1] = x1; jinfo[2] = y2; jinfo[3] = x2;
        jinfo[4] = (y2 - y1) * (x2 - x1);
        jinfo[5] = fv;
        jint[0] = ccls[o];
        jint[1] = fi;
        jint[2] = cidx[o];
        sc[fi] = -__builtin_inff();
      } else {
        jint[1] = -1;
      }
    }
    __syncthreads();

    const int  jbi   = jint[1];
    const bool valid = (jbi >= 0);
    const float tag  = valid ? 1.0f : 0.0f;
    const size_t rk  = (size_t)b * KK + (size_t)k;

    // ---- emit output row k (zero-tagged when exhausted)
    if (t < 82) {
      float v;
      if (t == 81) v = tag;
      else         v = valid ? logits[((size_t)b * NN + (size_t)jint[2]) * CC + t] : 0.0f;
      ologits[rk * 82 + t] = v;
    } else if (t < 87) {
      const int c5 = t - 82;
      oboxes[rk * 5 + c5] = (c5 == 4) ? tag : (valid ? jinfo[c5] : 0.0f);
    } else if (t == 87) {
      oscores[rk * 2 + 0] = valid ? jinfo[5] : 0.0f;
    } else if (t == 88) {
      oscores[rk * 2 + 1] = tag;
    } else if (t == 89) {
      oclsout[rk * 2 + 0] = valid ? (float)jint[0] : 0.0f;
    } else if (t == 90) {
      oclsout[rk * 2 + 1] = tag;
    }

    // ---- suppress same-class overlaps
    if (valid) {
      const float jy1 = jinfo[0], jx1 = jinfo[1], jy2 = jinfo[2], jx2 = jinfo[3];
      const float jarea = jinfo[4];
      const int   jc    = jint[0];
      for (int i = t; i < M; i += 256) {
        if (i + 256 < M) {
          __builtin_prefetch(&cy1[base + i + 256], 0, 3);   // global_prefetch_b8
          __builtin_prefetch(&cy2[base + i + 256], 0, 3);
        }
        if (sc[i] > -__builtin_inff() && (int)scls8[i] == jc) {
          const float y1 = cy1[base + i], x1 = cx1[base + i];
          const float y2 = cy2[base + i], x2 = cx2[base + i];
          const float iy = fmaxf(0.0f, fminf(jy2, y2) - fmaxf(jy1, y1));
          const float ix = fmaxf(0.0f, fminf(jx2, x2) - fmaxf(jx1, x1));
          const float inter = iy * ix;
          const float area  = (y2 - y1) * (x2 - x1);
          const float iou   = inter / (jarea + area - inter);
          if (iou > IOU_THR) sc[i] = -__builtin_inff();
        }
      }
    }
    __syncthreads();
  }
}

// ---------------------------------------------------------------------------
extern "C" void kernel_launch(void* const* d_in, const int* in_sizes, int n_in,
                              void* d_out, int out_size, void* d_ws, size_t ws_size,
                              hipStream_t stream) {
  const float* deltas = (const float*)d_in[0];   // [32,8000,4]
  const float* clogit = (const float*)d_in[1];   // [32,8000,81]
  const float* props  = (const float*)d_in[2];   // [32,8000,5]
  float* out = (float*)d_out;

  const size_t BN = (size_t)BB * NN;
  char* w = (char*)d_ws;
  int*   ccount = (int*)w;                       // 32 ints (256B slot)
  float* cy1    = (float*)(w + 256);
  float* cx1    = cy1 + BN;
  float* cy2    = cx1 + BN;
  float* cx2    = cy2 + BN;
  float* cscore = cx2 + BN;
  int*   ccls   = (int*)(cscore + BN);
  int*   cidx   = ccls + BN;
  (void)in_sizes; (void)n_in; (void)out_size; (void)ws_size;

  init_counts_kernel<<<1, 32, 0, stream>>>(ccount);

  // 16 proposals/wave, 8 waves/block -> 128 proposals/block; 256000/128 = 2000
  score_compact_kernel<<<2000, 256, 0, stream>>>(
      deltas, clogit, props, cy1, cx1, cy2, cx2, cscore, ccls, cidx, ccount);

  nms_kernel<<<BB, 256, 0, stream>>>(
      clogit, cy1, cx1, cy2, cx2, cscore, ccls, cidx, ccount, out);
}
